// Attention_63754494542295
// MI455X (gfx1250) — compile-verified, matrix operand order checked by reading, not run on hardware
//
#include <hip/hip_runtime.h>

// ---------------------------------------------------------------------------
// Problem constants (from the reference)
// ---------------------------------------------------------------------------
#define BB 4
#define NN 2048
#define DIMM 512
#define HEADS 8
#define DHEAD 64
#define ROWS (BB * NN)          // 8192
#define SCALE 0.125f            // 64^-0.5
#define EPS 1e-5f

typedef __attribute__((ext_vector_type(16))) __bf16 v16bf;
typedef __attribute__((ext_vector_type(8)))  float  v8f;

union TileU { unsigned u[8]; v16bf v; };

// ---------------------------------------------------------------------------
// Helpers
// ---------------------------------------------------------------------------
__device__ __forceinline__ unsigned short f2bf(float f) {
    unsigned u = __float_as_uint(f);
    unsigned r = u + 0x7FFFu + ((u >> 16) & 1u);   // round-to-nearest-even
    return (unsigned short)(r >> 16);
}
__device__ __forceinline__ unsigned pack2bf(float a, float b) {
    return (unsigned)f2bf(a) | ((unsigned)f2bf(b) << 16);
}

// Load a 16x32 bf16 tile in WMMA A-layout (also B-layout when source is
// K-major). Lane l supplies row (l&15); dword v holds K = 16*(v>>2) + 8*(l>>4)
// + 2*(v&3) .. +1.  The 8 dwords are two contiguous b128 loads.
__device__ __forceinline__ void load_tile(const unsigned short* __restrict__ base,
                                          int ld, unsigned out[8]) {
    int lane = threadIdx.x & 31;
    const unsigned short* row = base + (size_t)(lane & 15) * ld + ((lane >> 4) << 3);
    uint4 lo = *reinterpret_cast<const uint4*>(row);
    uint4 hi = *reinterpret_cast<const uint4*>(row + 16);
    out[0] = lo.x; out[1] = lo.y; out[2] = lo.z; out[3] = lo.w;
    out[4] = hi.x; out[5] = hi.y; out[6] = hi.z; out[7] = hi.w;
}

__device__ __forceinline__ v8f bf16_wmma(const unsigned a[8], const unsigned b[8], v8f c) {
    TileU ta, tb;
#pragma unroll
    for (int i = 0; i < 8; ++i) { ta.u[i] = a[i]; tb.u[i] = b[i]; }
    return __builtin_amdgcn_wmma_f32_16x16x32_bf16(
        /*neg_a=*/false, ta.v, /*neg_b=*/false, tb.v,
        /*c_mod=*/(short)0, c, /*reuse_a=*/false, /*reuse_b=*/false);
}

__device__ __forceinline__ float rmax16(float x) {
#pragma unroll
    for (int m = 1; m < 16; m <<= 1) x = fmaxf(x, __shfl_xor(x, m, 16));
    return x;
}
__device__ __forceinline__ float rsum16(float x) {
#pragma unroll
    for (int m = 1; m < 16; m <<= 1) x += __shfl_xor(x, m, 16);
    return x;
}

// ---------------------------------------------------------------------------
// LayerNorm (one wave per 512-wide row), biased variance, gamma only
// ---------------------------------------------------------------------------
template <bool BF16OUT>
__global__ void ln_kernel(const float* __restrict__ in, const float* __restrict__ g,
                          void* __restrict__ out, int rows) {
    int wid  = threadIdx.x >> 5;
    int lane = threadIdx.x & 31;
    int row  = blockIdx.x * (blockDim.x >> 5) + wid;
    if (row >= rows) return;
    const float* xr = in + (size_t)row * DIMM;
    float vals[16], s = 0.f, sq = 0.f;
#pragma unroll
    for (int i = 0; i < 16; ++i) {
        float x = xr[lane + i * 32];
        vals[i] = x; s += x; sq += x * x;
    }
#pragma unroll
    for (int m = 1; m < 32; m <<= 1) { s += __shfl_xor(s, m, 32); sq += __shfl_xor(sq, m, 32); }
    float mean = s * (1.f / DIMM);
    float var  = sq * (1.f / DIMM) - mean * mean;
    float rstd = rsqrtf(var + EPS);
#pragma unroll
    for (int i = 0; i < 16; ++i) {
        int col = lane + i * 32;
        float y = (vals[i] - mean) * rstd * g[col];
        if (BF16OUT) ((unsigned short*)out)[(size_t)row * DIMM + col] = f2bf(y);
        else         ((float*)out)[(size_t)row * DIMM + col] = y;
    }
}

// ---------------------------------------------------------------------------
// Weight convert+transpose: fp32 [K,N] -> bf16 [N,K]
// ---------------------------------------------------------------------------
__global__ void wtrans_kernel(const float* __restrict__ w, unsigned short* __restrict__ wt,
                              int K, int N) {
    int idx = blockIdx.x * blockDim.x + threadIdx.x;
    if (idx >= K * N) return;
    int k = idx / N, n = idx % N;
    wt[(size_t)n * K + k] = f2bf(w[idx]);
}

// ---------------------------------------------------------------------------
// WMMA GEMM: A bf16 [M,512] row-major  x  Wt bf16 [Nout,512] (i.e. W^T)
// One wave computes a 16x64 output tile.  MODE selects the epilogue:
//   0: write Q bf16 [b,h,n,d] with SCALE folded in
//   1: write K bf16 [b,n,d] (cols 0..63) and V^T bf16 [b,d,n] (cols 64..127)
//   2: write fp32 row-major [M,512]
// ---------------------------------------------------------------------------
template <int MODE>
__global__ void gemm512_kernel(const unsigned short* __restrict__ A,
                               const unsigned short* __restrict__ Wt,
                               void* __restrict__ out0, void* __restrict__ out1,
                               int M, int Nout) {
    int wave = (int)((blockIdx.x * blockDim.x + threadIdx.x) >> 5);
    int nChunks = Nout >> 6;
    int nt = wave % nChunks;
    int mt = wave / nChunks;
    if (mt * 16 >= M) return;
    int lane = threadIdx.x & 31;
    int r = lane & 15, h = lane >> 4;

    v8f acc[4];
#pragma unroll
    for (int c = 0; c < 4; ++c)
#pragma unroll
        for (int v = 0; v < 8; ++v) acc[c][v] = 0.f;

    const unsigned short* arow = A + (size_t)mt * 16 * DIMM;
    for (int kk = 0; kk < DIMM; kk += 32) {
        unsigned at[8];
        load_tile(arow + kk, DIMM, at);
#pragma unroll
        for (int c = 0; c < 4; ++c) {
            unsigned bt[8];
            load_tile(Wt + (size_t)(nt * 64 + c * 16) * DIMM + kk, DIMM, bt);
            acc[c] = bf16_wmma(at, bt, acc[c]);
        }
    }
#pragma unroll
    for (int c = 0; c < 4; ++c)
#pragma unroll
        for (int v = 0; v < 8; ++v) {
            int m = mt * 16 + h * 8 + v;
            int n = nt * 64 + c * 16 + r;
            float val = acc[c][v];
            if (MODE == 0) {
                int b = m >> 11, i = m & (NN - 1), hh = n >> 6, d = n & 63;
                ((unsigned short*)out0)[(((size_t)(b * HEADS + hh) * NN) + i) * DHEAD + d] =
                    f2bf(val * SCALE);
            } else if (MODE == 1) {
                int b = m >> 11, i = m & (NN - 1);
                if (n < DHEAD)
                    ((unsigned short*)out0)[(size_t)m * DHEAD + n] = f2bf(val);
                else
                    ((unsigned short*)out1)[(size_t)(b * DHEAD + (n - DHEAD)) * NN + i] = f2bf(val);
            } else {
                ((float*)out0)[(size_t)m * DIMM + n] = val;
            }
        }
}

// ---------------------------------------------------------------------------
// Flash attention: one wave owns 16 query rows of one (b,h).
// Streams j in blocks of 32; sim accumulated in fp32 via 2x2 WMMAs, bias
// added, online softmax, P routed C-layout -> A-layout through LDS, then
// 4 WMMAs accumulate P @ V into a 16x64 fp32 tile.
// ---------------------------------------------------------------------------
__global__ void attn_kernel(const unsigned short* __restrict__ q,
                            const unsigned short* __restrict__ k,
                            const unsigned short* __restrict__ vt,
                            const float* __restrict__ bias,
                            unsigned short* __restrict__ aout) {
    __shared__ float plds[8][16 * 32];
    int wid  = threadIdx.x >> 5;
    int wave = blockIdx.x * 8 + wid;
    int it = wave & 127;                 // i-tile within (b,h)
    int hh = (wave >> 7) & (HEADS - 1);
    int b  = wave >> 10;
    int i0 = it * 16;
    int lane = threadIdx.x & 31;
    int r = lane & 15, hl = lane >> 4;

    unsigned qa0[8], qa1[8];
    const unsigned short* qbase = q + (((size_t)(b * HEADS + hh) * NN) + i0) * DHEAD;
    load_tile(qbase,      DHEAD, qa0);
    load_tile(qbase + 32, DHEAD, qa1);

    float mrun[8], lrun[8];
    v8f acc[4];
#pragma unroll
    for (int v = 0; v < 8; ++v) { mrun[v] = -3.0e38f; lrun[v] = 0.f; }
#pragma unroll
    for (int c = 0; c < 4; ++c)
#pragma unroll
        for (int v = 0; v < 8; ++v) acc[c][v] = 0.f;

    const float* bptr = bias + ((size_t)hh * NN + i0 + hl * 8) * NN;
    const unsigned short* kbatch = k + (size_t)b * NN * DHEAD;
    const unsigned short* vbatch = vt + (size_t)b * DHEAD * NN;

    for (int j0 = 0; j0 < NN; j0 += 32) {
        v8f sim[2];
#pragma unroll
        for (int t = 0; t < 2; ++t) {
            unsigned kb0[8], kb1[8];
            const unsigned short* kbase = kbatch + (size_t)(j0 + t * 16) * DHEAD;
            load_tile(kbase,      DHEAD, kb0);
            load_tile(kbase + 32, DHEAD, kb1);
            v8f c;
#pragma unroll
            for (int v = 0; v < 8; ++v) c[v] = 0.f;
            c = bf16_wmma(qa0, kb0, c);
            c = bf16_wmma(qa1, kb1, c);
#pragma unroll
            for (int v = 0; v < 8; ++v)
                c[v] += bptr[(size_t)v * NN + j0 + t * 16 + r];
            sim[t] = c;
        }
        float corr[8];
#pragma unroll
        for (int v = 0; v < 8; ++v) {
            float rm = rmax16(fmaxf(sim[0][v], sim[1][v]));
            float mn = fmaxf(mrun[v], rm);
            corr[v] = __expf(mrun[v] - mn);
            mrun[v] = mn;
        }
        float psum[8];
#pragma unroll
        for (int t = 0; t < 2; ++t)
#pragma unroll
            for (int v = 0; v < 8; ++v) {
                float p = __expf(sim[t][v] - mrun[v]);
                plds[wid][(hl * 8 + v) * 32 + t * 16 + r] = p;
                if (t == 0) psum[v] = p; else psum[v] += p;
            }
#pragma unroll
        for (int v = 0; v < 8; ++v)
            lrun[v] = lrun[v] * corr[v] + rsum16(psum[v]);
#pragma unroll
        for (int c4 = 0; c4 < 4; ++c4)
#pragma unroll
            for (int v = 0; v < 8; ++v) acc[c4][v] *= corr[v];

        // P: C-layout (in LDS) -> A-layout bf16 tile
        unsigned pa[8];
#pragma unroll
        for (int v = 0; v < 8; ++v) {
            int k0 = ((v >> 2) << 4) + (hl << 3) + ((v & 3) << 1);
            float f0 = plds[wid][r * 32 + k0];
            float f1 = plds[wid][r * 32 + k0 + 1];
            pa[v] = pack2bf(f0, f1);
        }
        // P @ V (V stored transposed -> B tiles are simple K-major loads)
#pragma unroll
        for (int c4 = 0; c4 < 4; ++c4) {
            unsigned vb[8];
            load_tile(vbatch + (size_t)(c4 * 16) * NN + j0, NN, vb);
            acc[c4] = bf16_wmma(pa, vb, acc[c4]);
        }
    }
    // epilogue: normalize and scatter to [b,n, h*64+d] bf16
#pragma unroll
    for (int c4 = 0; c4 < 4; ++c4)
#pragma unroll
        for (int v = 0; v < 8; ++v) {
            int m = i0 + hl * 8 + v;
            float o = acc[c4][v] / lrun[v];
            aout[((size_t)(b * NN + m)) * DIMM + hh * DHEAD + c4 * 16 + r] = f2bf(o);
        }
}

// ---------------------------------------------------------------------------
// Host-side launcher
// ---------------------------------------------------------------------------
extern "C" void kernel_launch(void* const* d_in, const int* in_sizes, int n_in,
                              void* d_out, int out_size, void* d_ws, size_t ws_size,
                              hipStream_t stream) {
    (void)in_sizes; (void)n_in; (void)out_size; (void)ws_size;
    const float* x     = (const float*)d_in[0];
    const float* bias  = (const float*)d_in[1];
    const float* w_q   = (const float*)d_in[2];
    const float* w_kv  = (const float*)d_in[3];
    const float* w_out = (const float*)d_in[4];
    const float* g_in  = (const float*)d_in[5];
    const float* g_out = (const float*)d_in[6];
    float* out = (float*)d_out;

    char* ws = (char*)d_ws;
    size_t off = 0;
    auto carve = [&](size_t bytes) -> void* {
        void* p = ws + off;
        off += (bytes + 255) & ~(size_t)255;
        return p;
    };
    unsigned short* xn   = (unsigned short*)carve((size_t)ROWS * DIMM * 2);         // LN'd input, bf16
    unsigned short* wqT  = (unsigned short*)carve((size_t)DIMM * DIMM * 2);         // w_q^T bf16
    unsigned short* wkvT = (unsigned short*)carve((size_t)128 * DIMM * 2);          // w_kv^T bf16
    unsigned short* woT  = (unsigned short*)carve((size_t)DIMM * DIMM * 2);         // w_out^T bf16
    unsigned short* qb   = (unsigned short*)carve((size_t)BB * HEADS * NN * DHEAD * 2);
    unsigned short* kb   = (unsigned short*)carve((size_t)BB * NN * DHEAD * 2);
    unsigned short* vtb  = (unsigned short*)carve((size_t)BB * DHEAD * NN * 2);
    unsigned short* ao   = (unsigned short*)carve((size_t)ROWS * DIMM * 2);
    float*          proj = (float*)carve((size_t)ROWS * DIMM * 4);

    // 1. input LayerNorm -> bf16
    ln_kernel<true><<<ROWS / 8, 256, 0, stream>>>(x, g_in, xn, ROWS);
    // 2. weight transposes (fp32 -> bf16, K-major)
    wtrans_kernel<<<(DIMM * DIMM) / 256, 256, 0, stream>>>(w_q, wqT, DIMM, DIMM);
    wtrans_kernel<<<(DIMM * 128) / 256, 256, 0, stream>>>(w_kv, wkvT, DIMM, 128);
    wtrans_kernel<<<(DIMM * DIMM) / 256, 256, 0, stream>>>(w_out, woT, DIMM, DIMM);
    // 3. projections (WMMA):  Q (scaled, [b,h,n,d]),  K ([b,n,d]) + V^T ([b,d,n])
    gemm512_kernel<0><<<512, 256, 0, stream>>>(xn, wqT, qb, nullptr, ROWS, DIMM);
    gemm512_kernel<1><<<128, 256, 0, stream>>>(xn, wkvT, kb, vtb, ROWS, 128);
    // 4. flash attention (WMMA QK^T + WMMA PV, streaming softmax + bias)
    attn_kernel<<<512, 256, 0, stream>>>(qb, kb, vtb, bias, ao);
    // 5. output projection (WMMA) -> fp32
    gemm512_kernel<2><<<512, 256, 0, stream>>>(ao, woT, proj, nullptr, ROWS, DIMM);
    // 6. output LayerNorm -> d_out
    ln_kernel<false><<<ROWS / 8, 256, 0, stream>>>(proj, g_out, out, ROWS);
}
